// HypergraphLearner_62826781606426
// MI455X (gfx1250) — matrix operand order repaired and checked by reading, not compiled, not run on hardware
//
#include <hip/hip_runtime.h>
#include <hip/hip_bf16.h>
#include <math.h>

// ---------------------------------------------------------------------------
// HypergraphLearner forward for MI455X (gfx1250).
// All GEMMs via v_wmma_f32_16x16x32_f16 (f16 A/B, f32 accumulate), wave32.
// Each wave register-blocks a 32x32 output tile: 4 independent WMMAs per
// K-step.  GEMM is templated on K-contiguity of A and B so the hot paths use
// vectorized float4 (global_load_b128) fragment loads instead of 64 scalar
// loads per K-step, plus global_prefetch of the next K-step.
// ---------------------------------------------------------------------------

typedef __attribute__((ext_vector_type(16))) _Float16 v16h;
typedef __attribute__((ext_vector_type(8)))  float    v8f;

#define GF_ACC   1   // Y += alpha*acc (bias/relu/resid ignored)
#define GF_RELU  2   // v = relu(alpha*acc + bias)
#define GF_RESID 4   // v += resid[m*ldy+n]

#define BB  2      // batch
#define DD  128    // model dim
#define HH  4      // heads
#define LL  64     // temporal hyperedges
#define EE  32     // variable hyperedges
#define NNN 2048   // tokens  (L*E)
#define NDC 2      // ND = D/L

__device__ inline v16h pack16(float4 x0, float4 x1, float4 x2, float4 x3)
{
  v16h r;
  r[0]  = (_Float16)x0.x; r[1]  = (_Float16)x0.y;
  r[2]  = (_Float16)x0.z; r[3]  = (_Float16)x0.w;
  r[4]  = (_Float16)x1.x; r[5]  = (_Float16)x1.y;
  r[6]  = (_Float16)x1.z; r[7]  = (_Float16)x1.w;
  r[8]  = (_Float16)x2.x; r[9]  = (_Float16)x2.y;
  r[10] = (_Float16)x2.z; r[11] = (_Float16)x2.w;
  r[12] = (_Float16)x3.x; r[13] = (_Float16)x3.y;
  r[14] = (_Float16)x3.z; r[15] = (_Float16)x3.w;
  return r;
}

// ---------------------------------------------------------------------------
// Generic WMMA GEMM:  Y[m,n] (+)= alpha * sum_k A(m,k)*B(k,n)  [+bias][relu][+resid]
// One wave (32 lanes) per 32x32 output tile; K stepped by 32; 2x2 register
// blocking -> 4 v_wmma per step.
// A(m,k) addr = Ab + m*sAm + k*sAk ; B(k,n) addr = Bb + k*sBk + n*sBn
// AKC: sAk==1 (A rows contiguous in K) -> float4 loads.
// BKC: sBk==1 (B cols contiguous in K) -> float4 loads.
// blockIdx.z decomposes as (outer,inner) for batch x heads.
// Fragment layouts per CDNA5 ISA 7.12.2 (wave32).
// Requires M,N,K multiples of 32 (true for all call sites).
// ---------------------------------------------------------------------------
template<bool AKC, bool BKC>
__global__ __launch_bounds__(32) void wmma_gemm_kernel(
    const float* __restrict__ A, const float* __restrict__ Bm,
    const float* __restrict__ bias, const float* __restrict__ resid,
    float* __restrict__ Y,
    int M, int N, int K,
    int sAm, int sAk, long bsA, long bsA2,
    int sBk, int sBn, long bsB, long bsB2,
    int ldy, long bsY, long bsY2,
    long bsR, long bsR2,
    int inner, float alpha, int flags)
{
  (void)M; (void)N;
  int lane  = threadIdx.x;
  int z     = blockIdx.z;
  int bz    = z / inner;
  int iz    = z - bz * inner;
  const float* Ab = A + bz * bsA + iz * bsA2;
  const float* Bb = Bm + bz * bsB + iz * bsB2;
  float*       Yb = Y + bz * bsY + iz * bsY2;
  const float* Rb = resid ? (resid + bz * bsR + iz * bsR2) : (const float*)0;

  int half   = lane >> 4;        // 0 | 1
  int lane16 = lane & 15;
  int rowA0  = blockIdx.x * 32 + lane16;   // A rows this lane feeds
  int rowA1  = rowA0 + 16;
  int colB0  = blockIdx.y * 32 + lane16;   // B cols this lane feeds
  int colB1  = colB0 + 16;
  int offA   = half * 8;                   // A K-offset per lane half
  int kB0    = half * 8;                   // B K-offset per lane half

  // loop-invariant base pointers (K advances by +32 per step)
  const float* pA0 = Ab + (long)rowA0 * sAm;
  const float* pA1 = Ab + (long)rowA1 * sAm;
  const float* pB0 = Bb + (long)colB0 * sBn;
  const float* pB1 = Bb + (long)colB1 * sBn;

  v8f acc00 = {}, acc01 = {}, acc10 = {}, acc11 = {};

#pragma unroll 2
  for (int k0 = 0; k0 < K; k0 += 32) {
    v16h a0, a1, b0, b1;
    if (AKC) {
      // A fragment = two contiguous 8-float runs at K = k0+offA and +16
      const float4* q0 = (const float4*)(pA0 + k0 + offA);
      const float4* q1 = (const float4*)(pA1 + k0 + offA);
      a0 = pack16(q0[0], q0[1], q0[4], q0[5]);   // +16 floats == +4 float4
      a1 = pack16(q1[0], q1[1], q1[4], q1[5]);
    } else {
#pragma unroll
      for (int j = 0; j < 16; ++j) {
        int ka  = k0 + offA + j + ((j >= 8) ? 8 : 0);
        long ko = (long)ka * sAk;
        a0[j] = (_Float16)pA0[ko];
        a1[j] = (_Float16)pA1[ko];
      }
    }
    if (BKC) {
      // B fragment = two contiguous 8-float runs at K = k0+kB0 and +16
      const float4* q0 = (const float4*)(pB0 + k0 + kB0);
      const float4* q1 = (const float4*)(pB1 + k0 + kB0);
      b0 = pack16(q0[0], q0[1], q0[4], q0[5]);
      b1 = pack16(q1[0], q1[1], q1[4], q1[5]);
    } else {
#pragma unroll
      for (int j = 0; j < 16; ++j) {
        long ko = (long)(k0 + kB0 + j + ((j >= 8) ? 8 : 0)) * sBk;
        b0[j] = (_Float16)pB0[ko];
        b1[j] = (_Float16)pB1[ko];
      }
    }
    if (k0 + 32 < K) {
      // hide next fragment fetch behind the 4 WMMAs (global_prefetch_b8)
      __builtin_prefetch(&pA0[(long)(k0 + 32) * (AKC ? 1 : sAk)], 0, 1);
      __builtin_prefetch(&pA1[(long)(k0 + 32) * (AKC ? 1 : sAk)], 0, 1);
      __builtin_prefetch(&pB0[(long)(k0 + 32) * (BKC ? 1 : sBk)], 0, 1);
      __builtin_prefetch(&pB1[(long)(k0 + 32) * (BKC ? 1 : sBk)], 0, 1);
    }
    // 4 independent matrix ops -> XDL pipe ILP, no D->A/B hazards
    acc00 = __builtin_amdgcn_wmma_f32_16x16x32_f16(false, a0, false, b0,
                                                   (short)0, acc00, false, false);
    acc01 = __builtin_amdgcn_wmma_f32_16x16x32_f16(false, a0, false, b1,
                                                   (short)0, acc01, false, false);
    acc10 = __builtin_amdgcn_wmma_f32_16x16x32_f16(false, a1, false, b0,
                                                   (short)0, acc10, false, false);
    acc11 = __builtin_amdgcn_wmma_f32_16x16x32_f16(false, a1, false, b1,
                                                   (short)0, acc11, false, false);
  }

  auto writeback = [&](const v8f& acc, int mBase, int n) {
#pragma unroll
    for (int r = 0; r < 8; ++r) {
      // D layout: VGPR r -> row (r + 8*half), col = lane16
      int  m   = mBase + r + half * 8;
      long idx = (long)m * ldy + n;
      float v  = alpha * acc[r];
      if (flags & GF_ACC) { Yb[idx] += v; continue; }
      if (bias) v += bias[n];
      if (flags & GF_RELU)  v = v > 0.f ? v : 0.f;
      if (flags & GF_RESID) v += Rb[idx];
      Yb[idx] = v;
    }
  };
  writeback(acc00, blockIdx.x * 32,      colB0);
  writeback(acc01, blockIdx.x * 32,      colB1);
  writeback(acc10, blockIdx.x * 32 + 16, colB0);
  writeback(acc11, blockIdx.x * 32 + 16, colB1);
}

// ---------------------------------------------------------------------------
// Row softmax with optional masking (in-place on S, rows of length nk).
// mode 0: none; 1: mask[b,q,k]==0 -> -1e9; 2: xymf[b,q]*xymf[b,k]==0 -> -1e9
// ---------------------------------------------------------------------------
__global__ __launch_bounds__(256) void softmax_rows_kernel(
    float* __restrict__ S, int nk, int H, int nq, int mode,
    const float* __restrict__ mask, long mBs, long mQs,
    const float* __restrict__ xymf)
{
  __shared__ float red[256];
  int row = blockIdx.x;
  int t   = threadIdx.x;
  int b   = row / (H * nq);
  int q   = row % nq;
  float* Srow = S + (long)row * nk;

  float mq = (mode == 2) ? xymf[(long)b * nq + q] : 0.f;
  float mx = -3.0e38f;
  for (int k = t; k < nk; k += 256) {
    float v = Srow[k];
    if (mode == 1) {
      if (mask[mBs * b + mQs * q + k] == 0.f) v = -1.0e9f;
    } else if (mode == 2) {
      if (mq * xymf[(long)b * nk + k] == 0.f) v = -1.0e9f;
    }
    Srow[k] = v;
    mx = fmaxf(mx, v);
  }
  red[t] = mx; __syncthreads();
  for (int s = 128; s > 0; s >>= 1) {
    if (t < s) red[t] = fmaxf(red[t], red[t + s]);
    __syncthreads();
  }
  mx = red[0]; __syncthreads();

  float sum = 0.f;
  for (int k = t; k < nk; k += 256) {
    float e = __expf(Srow[k] - mx);
    Srow[k] = e;
    sum += e;
  }
  red[t] = sum; __syncthreads();
  for (int s = 128; s > 0; s >>= 1) {
    if (t < s) red[t] += red[t + s];
    __syncthreads();
  }
  float inv = 1.f / red[0];
  for (int k = t; k < nk; k += 256) Srow[k] *= inv;
}

// out[b,n,:] = src[b, idx[b,n], :]
__global__ void gather_rows_kernel(float* __restrict__ out,
                                   const float* __restrict__ src,
                                   const int* __restrict__ idx,
                                   int n, int d, int srcRows, long total)
{
  long i = (long)blockIdx.x * blockDim.x + threadIdx.x;
  if (i >= total) return;
  int dd  = (int)(i % d);
  long r  = i / d;
  int nn_ = (int)(r % n);
  int b   = (int)(r / n);
  int row = idx[(long)b * n + nn_];
  out[i]  = src[((long)b * srcRows + row) * d + dd];
}

// layer-0 masks: out[b,l,n] = inc[b,l,n] * g(1 - ymlf[b,n]),  g(0)=1e-8
__global__ void build_mask_l0_kernel(float* __restrict__ out,
                                     const float* __restrict__ inc,
                                     const float* __restrict__ ymlf,
                                     int rows, long total)
{
  long i = (long)blockIdx.x * blockDim.x + threadIdx.x;
  if (i >= total) return;
  int nn_ = (int)(i % NNN);
  long r  = i / NNN;
  int b   = (int)(r / rows);
  float mt = 1.f - ymlf[(long)b * NNN + nn_];
  if (mt == 0.f) mt = 1e-8f;
  out[i] = inc[i] * mt;
}

// y += a
__global__ void addinto_kernel(float* __restrict__ y, const float* __restrict__ a, long total)
{
  long i = (long)blockIdx.x * blockDim.x + threadIdx.x;
  if (i < total) y[i] += a[i];
}

// obs = relu((obs + lin + tanh(gate)*resid) * xymf[b,n])
__global__ void qsf_final_kernel(float* __restrict__ obs,
                                 const float* __restrict__ lin,
                                 const float* __restrict__ resid,
                                 const float* __restrict__ gate,
                                 const float* __restrict__ xymf, long total)
{
  long i = (long)blockIdx.x * blockDim.x + threadIdx.x;
  if (i >= total) return;
  long r  = i / DD;                 // b*N + n
  float g = tanhf(gate[0]);
  float v = (obs[i] + lin[i] + g * resid[i]) * xymf[r];
  obs[i]  = v > 0.f ? v : 0.f;
}

// global cumsum-compaction of obs[:,:, :ND] through the xym mask, then
// transpose (B,L,E,ND)->(B,E,L*ND).  Single block, LDS prefix scan.
__global__ __launch_bounds__(256) void fine_grained_kernel(
    float* __restrict__ qk, const float* __restrict__ obs,
    const float* __restrict__ xym)
{
  const int T      = BB * LL * EE * NDC;   // 8192
  const int chunk  = T / 256;              // 32
  const int srcLen = BB * NNN * NDC;       // 8192
  __shared__ int pref[257];
  int t    = threadIdx.x;
  int base = t * chunk;

  int cnt = 0;
  for (int j = 0; j < chunk; ++j) {
    int i = base + j;
    int e = (i / NDC) % EE;
    int l = (i / (NDC * EE)) % LL;
    int b = i / (NDC * EE * LL);
    if (xym[((long)b * LL + l) * EE + e] > 0.f) cnt++;
  }
  pref[t + 1] = cnt;
  __syncthreads();
  if (t == 0) { pref[0] = 0; for (int j = 1; j <= 256; ++j) pref[j] += pref[j - 1]; }
  __syncthreads();

  int pos = pref[t];
  for (int j = 0; j < chunk; ++j) {
    int i = base + j;
    int d = i % NDC;
    int e = (i / NDC) % EE;
    int l = (i / (NDC * EE)) % LL;
    int b = i / (NDC * EE * LL);
    float val = 0.f;
    if (xym[((long)b * LL + l) * EE + e] > 0.f) {
      int p  = pos < srcLen - 1 ? pos : srcLen - 1;
      int sb = p / (NNN * NDC);
      int rr = p % (NNN * NDC);
      int sn = rr / NDC;
      int sd = rr % NDC;
      val = obs[((long)sb * NNN + sn) * DD + sd];
      pos++;
    }
    qk[((long)b * EE + e) * (LL * NDC) + l * NDC + d] = val;
  }
}

// per-row diagonal normalization of mc[b,e,:]
__global__ void mc_norm_kernel(float* __restrict__ mc)
{
  int i = blockIdx.x * blockDim.x + threadIdx.x;
  if (i >= BB * EE) return;
  int b = i / EE, e = i % EE;
  float* row = mc + ((long)b * EE + e) * EE;
  float nop = row[e];
  if (nop != 0.f) {
    for (int f = 0; f < EE; ++f) row[f] /= nop;
  }
}

// scores = where((scores>thr)&(saux!=0), (1-mc)*scores + mc*saux, scores)
__global__ void iaa_mix_kernel(float* __restrict__ sc,
                               const float* __restrict__ saux,
                               const float* __restrict__ mc,
                               const float* __restrict__ thr, long total)
{
  long i = (long)blockIdx.x * blockDim.x + threadIdx.x;
  if (i >= total) return;
  float s = sc[i], sa = saux[i], m = mc[i];
  if (s > thr[0] && sa != 0.f) sc[i] = (1.f - m) * s + m * sa;
}

// ---------------------------------------------------------------------------
// host side
// ---------------------------------------------------------------------------
static inline void gemm_launch(hipStream_t st,
    const float* A, const float* Bm, const float* bias, const float* resid, float* Y,
    int M, int N, int K,
    int sAm, int sAk, long bsA, long bsA2,
    int sBk, int sBn, long bsB, long bsB2,
    int ldy, long bsY, long bsY2, long bsR, long bsR2,
    int outer, int inner, float alpha, int flags)
{
  dim3 g((unsigned)(M / 32), (unsigned)(N / 32), (unsigned)(outer * inner));
  bool akc = (sAk == 1);
  bool bkc = (sBk == 1);
  if (akc && bkc)
    wmma_gemm_kernel<true, true><<<g, 32, 0, st>>>(A, Bm, bias, resid, Y, M, N, K,
        sAm, sAk, bsA, bsA2, sBk, sBn, bsB, bsB2, ldy, bsY, bsY2, bsR, bsR2, inner, alpha, flags);
  else if (akc)
    wmma_gemm_kernel<true, false><<<g, 32, 0, st>>>(A, Bm, bias, resid, Y, M, N, K,
        sAm, sAk, bsA, bsA2, sBk, sBn, bsB, bsB2, ldy, bsY, bsY2, bsR, bsR2, inner, alpha, flags);
  else if (bkc)
    wmma_gemm_kernel<false, true><<<g, 32, 0, st>>>(A, Bm, bias, resid, Y, M, N, K,
        sAm, sAk, bsA, bsA2, sBk, sBn, bsB, bsB2, ldy, bsY, bsY2, bsR, bsR2, inner, alpha, flags);
  else
    wmma_gemm_kernel<false, false><<<g, 32, 0, st>>>(A, Bm, bias, resid, Y, M, N, K,
        sAm, sAk, bsA, bsA2, sBk, sBn, bsB, bsB2, ldy, bsY, bsY2, bsR, bsR2, inner, alpha, flags);
}

struct MabP { const float *qW,*qb,*kW,*kb,*vW,*vb,*oW,*ob; };

extern "C" void kernel_launch(void* const* d_in, const int* in_sizes, int n_in,
                              void* d_out, int out_size, void* d_ws, size_t ws_size,
                              hipStream_t stream)
{
  (void)in_sizes; (void)n_in; (void)out_size; (void)ws_size;
  auto F = [&](int i) { return (const float*)d_in[i]; };

  // ---- input map (setup_inputs order; params pytree in sorted-key order) ----
  const float* in_obs  = F(0);
  const float* in_th   = F(1);
  const float* in_vh   = F(2);
  const int*   tidx    = (const int*)d_in[3];
  const int*   vidx    = (const int*)d_in[4];
  const float* tim     = F(5);
  const float* vim     = F(6);
  const float* xymf    = F(7);
  const float* xym     = F(8);
  const float* ymlf    = F(9);
  // params: iaa{k(W,b),q(W,b),thr,v(W,b)}=10..16,
  // n2t0=17,n2t1=25,n2v0=33,n2v1=41,nsu0=49,nsu1=57 each {k,o,q,v}(W,b),
  // qsf0=65,qsf1=75 each {Pi,Pj,Pk,Wi,Wj,Wk,Wr,gate,linW,linb}
  const float* iaa_kW = F(10); const float* iaa_kb = F(11);
  const float* iaa_qW = F(12); const float* iaa_qb = F(13);
  const float* iaa_thr= F(14);
  const float* iaa_vW = F(15); const float* iaa_vb = F(16);
  auto mabp = [&](int base) {
    MabP m;
    m.kW = F(base + 0); m.kb = F(base + 1);
    m.oW = F(base + 2); m.ob = F(base + 3);
    m.qW = F(base + 4); m.qb = F(base + 5);
    m.vW = F(base + 6); m.vb = F(base + 7);
    return m;
  };
  const int n2t_base[2] = {17, 25};
  const int n2v_base[2] = {33, 41};
  const int nsu_base[2] = {49, 57};
  const int qsf_base[2] = {65, 75};

  // ---- workspace layout (floats) ----
  float* W = (float*)d_ws;
  size_t off = 0;
  auto alloc = [&](size_t n) { float* p = W + off; off += n; return p; };
  const long ND_  = (long)BB * NNN * DD;  // 524288
  float* obs   = alloc(ND_);
  float* th    = alloc((long)BB * LL * DD);
  float* vh    = alloc((long)BB * EE * DD);
  float* thn   = alloc((long)BB * LL * DD);
  float* vhn   = alloc((long)BB * EE * DD);
  float* bufA  = alloc(ND_);   // vg
  float* bufB  = alloc(ND_);   // tg
  float* qbuf  = alloc(ND_);
  float* kbuf  = alloc(ND_);
  float* vbuf  = alloc(ND_);
  float* obuf  = alloc(ND_);
  float* hbuf  = alloc(ND_);   // obs_h
  float* linb  = alloc(ND_);
  float* resb  = alloc(ND_);
  float* qib   = alloc(ND_);
  float* qjb   = alloc(ND_);
  float* qkb   = alloc(ND_);
  float* maskT = alloc((long)BB * LL * NNN);
  float* maskV = alloc((long)BB * EE * NNN);
  float* score = alloc((long)BB * HH * NNN * NNN);  // 134 MB, biggest buffer
  float* qkfg  = alloc((long)BB * EE * LL * NDC);
  float* mcb   = alloc((long)BB * EE * EE);
  float* saux  = alloc((long)BB * EE * EE);
  float* iasc  = alloc((long)BB * EE * EE);
  float* iaq   = alloc((long)BB * EE * DD);
  float* iak   = alloc((long)BB * EE * DD);
  float* iav   = alloc((long)BB * EE * DD);

  const float inv_sqrtD = 1.0f / sqrtf((float)DD);
  auto blk = [](long total) { return (unsigned)((total + 255) / 256); };

  // ---- init mutable state ----
  hipMemcpyAsync(obs, in_obs, ND_ * 4, hipMemcpyDeviceToDevice, stream);
  hipMemcpyAsync(th,  in_th,  (long)BB * LL * DD * 4, hipMemcpyDeviceToDevice, stream);
  hipMemcpyAsync(vh,  in_vh,  (long)BB * EE * DD * 4, hipMemcpyDeviceToDevice, stream);

  // ---- MAB: dest = O + relu(lin(O)),  O = q + attn(q, k, v) ----
  auto run_mab = [&](const float* Qin, int nq,
                     const float* s1, const float* s2, const float* s3,
                     const MabP& p, int KW, float* dest,
                     int maskMode, const float* maskPtr, int maskRows)
  {
    long bq = (long)nq * DD;
    // q = lin(Q)
    gemm_launch(stream, Qin, p.qW, p.qb, nullptr, qbuf, nq, DD, DD,
                DD, 1, bq, 0,  1, DD, 0, 0,  DD, bq, 0, 0, 0, BB, 1, 1.f, 0);
    // k/v = lin(concat parts) via sliced-weight accumulation
    const float* parts[3] = {s1, s2, s3};
    int np = s3 ? 3 : 2;
    for (int pi = 0; pi < np; ++pi) {
      int fl = pi == 0 ? 0 : GF_ACC;
      gemm_launch(stream, parts[pi], p.kW + pi * DD, pi == 0 ? p.kb : nullptr, nullptr,
                  kbuf, NNN, DD, DD, DD, 1, (long)NNN * DD, 0,
                  1, KW, 0, 0, DD, (long)NNN * DD, 0, 0, 0, BB, 1, 1.f, fl);
      gemm_launch(stream, parts[pi], p.vW + pi * DD, pi == 0 ? p.vb : nullptr, nullptr,
                  vbuf, NNN, DD, DD, DD, 1, (long)NNN * DD, 0,
                  1, KW, 0, 0, DD, (long)NNN * DD, 0, 0, 0, BB, 1, 1.f, fl);
    }
    // scores[b,h,q,kk] = qh . kh / sqrt(D)
    gemm_launch(stream, qbuf, kbuf, nullptr, nullptr, score, nq, NNN, 32,
                DD, 1, bq, 32,  1, DD, (long)NNN * DD, 32,
                NNN, (long)HH * nq * NNN, (long)nq * NNN, 0, 0,
                BB, HH, inv_sqrtD, 0);
    // mask + softmax
    softmax_rows_kernel<<<(unsigned)(BB * HH * nq), 256, 0, stream>>>(
        score, NNN, HH, nq, maskMode, maskPtr,
        (long)maskRows * NNN, (long)NNN, xymf);
    // obuf[b,q,h*32+d] = P @ v_h
    gemm_launch(stream, score, vbuf, nullptr, nullptr, obuf, nq, 32, NNN,
                NNN, 1, (long)HH * nq * NNN, (long)nq * NNN,
                DD, 1, (long)NNN * DD, 32,
                DD, bq, 32, 0, 0, BB, HH, 1.f, 0);
    // O = q + o
    addinto_kernel<<<blk((long)BB * bq), 256, 0, stream>>>(obuf, qbuf, (long)BB * bq);
    // dest = O + relu(O @ Wo^T + bo)
    gemm_launch(stream, obuf, p.oW, p.ob, obuf, dest, nq, DD, DD,
                DD, 1, bq, 0,  1, DD, 0, 0,
                DD, bq, 0, bq, 0, BB, 1, 1.f, GF_RELU | GF_RESID);
  };

  // =========================== layers ===========================
  for (int i = 0; i < 2; ++i) {
    MabP n2t = mabp(n2t_base[i]);
    MabP n2v = mabp(n2v_base[i]);
    MabP nsu = mabp(nsu_base[i]);
    int qb_  = qsf_base[i];
    const float* Pi = F(qb_ + 0); const float* Pj = F(qb_ + 1); const float* Pk = F(qb_ + 2);
    const float* Wi = F(qb_ + 3); const float* Wj = F(qb_ + 4); const float* Wk = F(qb_ + 5);
    const float* Wr = F(qb_ + 6); const float* gate = F(qb_ + 7);
    const float* linW = F(qb_ + 8); const float* linBias = F(qb_ + 9);

    // gathers of old hyperedge states
    gather_rows_kernel<<<blk(ND_), 256, 0, stream>>>(bufA, vh, vidx, NNN, DD, EE, ND_);
    gather_rows_kernel<<<blk(ND_), 256, 0, stream>>>(bufB, th, tidx, NNN, DD, LL, ND_);

    const float* mT = tim;
    const float* mV = vim;
    if (i == 0) {
      build_mask_l0_kernel<<<blk((long)BB * LL * NNN), 256, 0, stream>>>(
          maskT, tim, ymlf, LL, (long)BB * LL * NNN);
      build_mask_l0_kernel<<<blk((long)BB * EE * NNN), 256, 0, stream>>>(
          maskV, vim, ymlf, EE, (long)BB * EE * NNN);
      mT = maskT; mV = maskV;
    }

    run_mab(th, LL, bufA, obs, nullptr, n2t, 2 * DD, thn, 1, mT, LL);
    run_mab(vh, EE, bufB, obs, nullptr, n2v, 2 * DD, vhn, 1, mV, EE);
    hipMemcpyAsync(th, thn, (long)BB * LL * DD * 4, hipMemcpyDeviceToDevice, stream);
    hipMemcpyAsync(vh, vhn, (long)BB * EE * DD * 4, hipMemcpyDeviceToDevice, stream);

    // gathers of new states
    gather_rows_kernel<<<blk(ND_), 256, 0, stream>>>(bufB, th, tidx, NNN, DD, LL, ND_); // tg
    gather_rows_kernel<<<blk(ND_), 256, 0, stream>>>(bufA, vh, vidx, NNN, DD, EE, ND_); // vg

    // nsu self-attention: Q=obs, K=concat([tg, vg, obs]), outer-product mask
    run_mab(obs, NNN, bufB, bufA, obs, nsu, 3 * DD, hbuf, 2, nullptr, 0);

    // ---- qsf ----
    // lin = concat([obs_h, tg, vg]) @ linW^T + b  (sliced-weight accumulation)
    const float* qparts[3] = {hbuf, bufB, bufA};
    for (int pi = 0; pi < 3; ++pi) {
      gemm_launch(stream, qparts[pi], linW + pi * DD, pi == 0 ? linBias : nullptr, nullptr,
                  linb, NNN, DD, DD, DD, 1, ND_ / BB, 0,
                  1, 3 * DD, 0, 0, DD, ND_ / BB, 0, 0, 0,
                  BB, 1, 1.f, pi == 0 ? 0 : GF_ACC);
    }
    // qi/qj/qk
    gemm_launch(stream, hbuf, Pi, nullptr, nullptr, qib, NNN, DD, DD,
                DD, 1, ND_ / BB, 0, 1, DD, 0, 0, DD, ND_ / BB, 0, 0, 0, BB, 1, 1.f, 0);
    gemm_launch(stream, bufB, Pj, nullptr, nullptr, qjb, NNN, DD, DD,
                DD, 1, ND_ / BB, 0, 1, DD, 0, 0, DD, ND_ / BB, 0, 0, 0, BB, 1, 1.f, 0);
    gemm_launch(stream, bufA, Pk, nullptr, nullptr, qkb, NNN, DD, DD,
                DD, 1, ND_ / BB, 0, 1, DD, 0, 0, DD, ND_ / BB, 0, 0, 0, BB, 1, 1.f, 0);
    // residual = lin@Wr^T - qi@Wi^T - qj@Wj^T - qk@Wk^T
    gemm_launch(stream, linb, Wr, nullptr, nullptr, resb, NNN, DD, DD,
                DD, 1, ND_ / BB, 0, 1, DD, 0, 0, DD, ND_ / BB, 0, 0, 0, BB, 1, 1.f, 0);
    gemm_launch(stream, qib, Wi, nullptr, nullptr, resb, NNN, DD, DD,
                DD, 1, ND_ / BB, 0, 1, DD, 0, 0, DD, ND_ / BB, 0, 0, 0, BB, 1, -1.f, GF_ACC);
    gemm_launch(stream, qjb, Wj, nullptr, nullptr, resb, NNN, DD, DD,
                DD, 1, ND_ / BB, 0, 1, DD, 0, 0, DD, ND_ / BB, 0, 0, 0, BB, 1, -1.f, GF_ACC);
    gemm_launch(stream, qkb, Wk, nullptr, nullptr, resb, NNN, DD, DD,
                DD, 1, ND_ / BB, 0, 1, DD, 0, 0, DD, ND_ / BB, 0, 0, 0, BB, 1, -1.f, GF_ACC);
    // obs = relu((obs + h2n)*md), h2n = lin + tanh(gate)*residual
    qsf_final_kernel<<<blk(ND_), 256, 0, stream>>>(obs, linb, resb, gate, xymf, ND_);

    // ---- final-layer extras: fine_grained + iaa ----
    if (i == 1) {
      fine_grained_kernel<<<1, 256, 0, stream>>>(qkfg, obs, xym);
      // mc = xym^T @ xym  (per batch), then row-diagonal normalize
      gemm_launch(stream, xym, xym, nullptr, nullptr, mcb, EE, EE, LL,
                  1, EE, (long)LL * EE, 0,  EE, 1, (long)LL * EE, 0,
                  EE, (long)EE * EE, 0, 0, 0, BB, 1, 1.f, 0);
      mc_norm_kernel<<<1, 64, 0, stream>>>(mcb);
      // iaa q/k/v
      long bv = (long)EE * DD;
      gemm_launch(stream, vh, iaa_qW, iaa_qb, nullptr, iaq, EE, DD, DD,
                  DD, 1, bv, 0, 1, DD, 0, 0, DD, bv, 0, 0, 0, BB, 1, 1.f, 0);
      gemm_launch(stream, vh, iaa_kW, iaa_kb, nullptr, iak, EE, DD, DD,
                  DD, 1, bv, 0, 1, DD, 0, 0, DD, bv, 0, 0, 0, BB, 1, 1.f, 0);
      gemm_launch(stream, vh, iaa_vW, iaa_vb, nullptr, iav, EE, DD, DD,
                  DD, 1, bv, 0, 1, DD, 0, 0, DD, bv, 0, 0, 0, BB, 1, 1.f, 0);
      // scores = q @ k^T / sqrt(D) ; saux = qkfg @ qkfg^T / sqrt(128)
      gemm_launch(stream, iaq, iak, nullptr, nullptr, iasc, EE, EE, DD,
                  DD, 1, bv, 0, 1, DD, bv, 0,
                  EE, (long)EE * EE, 0, 0, 0, BB, 1, inv_sqrtD, 0);
      gemm_launch(stream, qkfg, qkfg, nullptr, nullptr, saux, EE, EE, DD,
                  DD, 1, bv, 0, 1, DD, bv, 0,
                  EE, (long)EE * EE, 0, 0, 0, BB, 1, inv_sqrtD, 0);
      iaa_mix_kernel<<<blk((long)BB * EE * EE), 256, 0, stream>>>(
          iasc, saux, mcb, iaa_thr, (long)BB * EE * EE);
      softmax_rows_kernel<<<(unsigned)(BB * EE), 256, 0, stream>>>(
          iasc, EE, 1, EE, 0, nullptr, 0, 0, nullptr);
      // vh = vh + P @ v
      gemm_launch(stream, iasc, iav, nullptr, vh, vhn, EE, DD, EE,
                  EE, 1, (long)EE * EE, 0,  DD, 1, bv, 0,
                  DD, bv, 0, bv, 0, BB, 1, 1.f, GF_RESID);
      hipMemcpyAsync(vh, vhn, bv * BB * 4, hipMemcpyDeviceToDevice, stream);
    }
  }

  // ---- outputs: concat(obs, th, vh) flat ----
  float* out = (float*)d_out;
  hipMemcpyAsync(out, obs, ND_ * 4, hipMemcpyDeviceToDevice, stream);
  hipMemcpyAsync(out + ND_, th, (long)BB * LL * DD * 4, hipMemcpyDeviceToDevice, stream);
  hipMemcpyAsync(out + ND_ + (long)BB * LL * DD, vh,
                 (long)BB * EE * DD * 4, hipMemcpyDeviceToDevice, stream);
}
